// ImitationNet_27281632264226
// MI455X (gfx1250) — matrix operand order repaired and checked by reading, not compiled, not run on hardware
//
#include <hip/hip_runtime.h>

// CDNA5 / gfx1250: wave32, WMMA bf16 16x16x32 (f32 accumulate),
// async global->LDS staging for the GEMM B operand, software-pipelined
// A fragments, batched single-buffered B fragments (no spills).

typedef __attribute__((ext_vector_type(16))) __bf16 v16bf;
typedef __attribute__((ext_vector_type(8)))  float  v8f;

#define NNODES 50000   // == 3125 * 16: every M tile is full
#define NEDGES 800000

union V16U { v16bf v; uint4 q[2]; };

__device__ __forceinline__ __bf16 f2bf(float f) { return (__bf16)f; }

// ---------------- degree / normalization ----------------
__global__ void deg_init(float* deg, int n) {
    int i = blockIdx.x * blockDim.x + threadIdx.x;
    if (i < n) deg[i] = 1.0f;  // self-loop
}
__global__ void deg_count(const int* __restrict__ col, float* deg, int e) {
    int i = blockIdx.x * blockDim.x + threadIdx.x;
    if (i < e) atomicAdd(&deg[col[i]], 1.0f);
}
__global__ void deg_rsqrt(const float* __restrict__ deg, float* dinv, int n) {
    int i = blockIdx.x * blockDim.x + threadIdx.x;
    if (i < n) dinv[i] = rsqrtf(deg[i]);
}

// ---------------- elementwise helpers ----------------
__global__ void f32_to_bf16(const float* __restrict__ in, __bf16* __restrict__ out, size_t n) {
    size_t i = (size_t)blockIdx.x * blockDim.x + threadIdx.x;
    if (i < n) out[i] = f2bf(in[i]);
}
__global__ void zero_f32(float* p, size_t n) {
    size_t i = (size_t)blockIdx.x * blockDim.x + threadIdx.x;
    if (i < n) p[i] = 0.0f;
}

// Pack W [K x Nout] (f32, row-major) into per-lane WMMA B layout (bf16):
// Bpk[((nt*KT + kt)*32 + lane)*16 + c] = W[(kt*32 + (c/8)*16 + (lane/16)*8 + c%8)*Nout + nt*16 + lane%16]
__global__ void pack_B(const float* __restrict__ W, __bf16* __restrict__ Bpk,
                       int K, int Nout, int NoutPad) {
    int idx = blockIdx.x * blockDim.x + threadIdx.x;
    int KT = K >> 5;
    int total = (NoutPad >> 4) * KT * 512;
    if (idx >= total) return;
    int c    = idx & 15;
    int lane = (idx >> 4) & 31;
    int t    = idx >> 9;
    int kt   = t % KT;
    int nt   = t / KT;
    int k = kt * 32 + ((c >> 3) << 4) + (lane >> 4) * 8 + (c & 7);
    int n = nt * 16 + (lane & 15);
    float v = (n < Nout) ? W[(size_t)k * Nout + n] : 0.0f;
    Bpk[idx] = f2bf(v);
}

// ---------------- WMMA GEMM, register-blocked 16 x (NTB*16) per wave ----------------
// Block = 8 waves covering 8 consecutive M tiles, all sharing one B slab of
// NTB*16 output columns x full K, staged in LDS via async global->LDS copies.
// A fragments are double-buffered (global loads in flight across the WMMA
// group); B fragments are batched into one ds_load clause per K-step.
// MODE 0: raw f32 out. MODE 1: bias+ReLU, bf16 out.
template<int NTB, int MODE, int KT>
__global__ void wmma_gemm_tiled(const __bf16* __restrict__ A,
                                const __bf16* __restrict__ Bpk,
                                const float* __restrict__ bias,
                                float* __restrict__ outF,
                                __bf16* __restrict__ outH,
                                int M, int NoutStore, int ngroups) {
    extern __shared__ char smemRaw[];
    const int K    = KT * 32;
    const int lane = threadIdx.x & 31;
    const int wv   = threadIdx.x >> 5;
    const int mg   = blockIdx.x / ngroups;
    const int ng   = blockIdx.x % ngroups;

    // ---- stage B slab (NTB*KT KiB) into LDS with async copies ----
    {
        const uint4* gB = (const uint4*)(Bpk + (size_t)(ng * NTB) * KT * 512);
        const int cnt = NTB * KT * 64;            // 16-byte chunks
        for (int i = threadIdx.x; i < cnt; i += blockDim.x) {
            unsigned int       ldsOff = (unsigned int)(unsigned long long)(smemRaw + (size_t)i * 16);
            unsigned long long gaddr  = (unsigned long long)(gB + i);
            asm volatile("global_load_async_to_lds_b128 %0, %1, off"
                         :: "v"(ldsOff), "v"(gaddr) : "memory");
        }
        asm volatile("s_wait_asynccnt 0x0" ::: "memory");
    }
    __syncthreads();

    const int  mtiles = M >> 4;
    int        mt     = mg * 8 + wv;
    const bool valid  = mt < mtiles;
    if (!valid) mt = mtiles - 1;                  // keep wave alive past barrier

    const int     row  = mt * 16 + (lane & 15);
    const __bf16* aRow = A + (size_t)row * K + ((lane >> 4) * 8);
    const char*   lB   = smemRaw + lane * 32;     // per-lane B base (16 bf16)

    v8f acc[NTB];
    const v8f vzero = {0.f, 0.f, 0.f, 0.f, 0.f, 0.f, 0.f, 0.f};
    #pragma unroll
    for (int j = 0; j < NTB; ++j) acc[j] = vzero;

    // prologue: A fragment for kt = 0
    V16U a;
    a.q[0] = *(const uint4*)(aRow);
    a.q[1] = *(const uint4*)(aRow + 16);

    #pragma unroll 2
    for (int kt = 0; kt < KT - 1; ++kt) {
        V16U b[NTB];
        #pragma unroll
        for (int j = 0; j < NTB; ++j) {           // one ds_load clause
            const char* bp = lB + (size_t)(j * KT + kt) * 1024;
            b[j].q[0] = *(const uint4*)(bp);
            b[j].q[1] = *(const uint4*)(bp + 16);
        }
        V16U an;                                  // next A fragment in flight
        {
            const __bf16* ap = aRow + (size_t)(kt + 1) * 32;
            an.q[0] = *(const uint4*)(ap);
            an.q[1] = *(const uint4*)(ap + 16);
        }
        if (kt + 4 < KT)                          // stream A ahead of use
            __builtin_prefetch(aRow + (size_t)(kt + 4) * 32, 0, 3);
        #pragma unroll
        for (int j = 0; j < NTB; ++j)             // NTB independent WMMAs, shared A
            acc[j] = __builtin_amdgcn_wmma_f32_16x16x32_bf16(false, a.v, false, b[j].v,
                                                             (short)0, acc[j], false, false);
        a = an;
    }
    {   // peeled last K-step (no next-A load, no branches)
        V16U b[NTB];
        #pragma unroll
        for (int j = 0; j < NTB; ++j) {
            const char* bp = lB + (size_t)(j * KT + (KT - 1)) * 1024;
            b[j].q[0] = *(const uint4*)(bp);
            b[j].q[1] = *(const uint4*)(bp + 16);
        }
        #pragma unroll
        for (int j = 0; j < NTB; ++j)
            acc[j] = __builtin_amdgcn_wmma_f32_16x16x32_bf16(false, a.v, false, b[j].v,
                                                             (short)0, acc[j], false, false);
    }

    if (!valid) return;
    // C/D: lanes 0-15 -> M = mt*16 + r ; lanes 16-31 -> M = mt*16 + 8 + r ; N = lane%16
    const int mbase = mt * 16 + ((lane >> 4) ? 8 : 0);
    #pragma unroll
    for (int j = 0; j < NTB; ++j) {
        const int n = (ng * NTB + j) * 16 + (lane & 15);
        if (n < NoutStore) {
            #pragma unroll
            for (int r = 0; r < 8; ++r) {
                const int m = mbase + r;
                float v = acc[j][r];
                if (MODE == 1) {
                    v = fmaxf(v + bias[n], 0.0f);
                    outH[(size_t)m * NoutStore + n] = f2bf(v);
                } else {
                    outF[(size_t)m * NoutStore + n] = v;
                }
            }
        }
    }
}

// ---------------- graph aggregation ----------------
__global__ void scatter_edges(const int* __restrict__ row, const int* __restrict__ col,
                              const float* __restrict__ dinv, const float* __restrict__ g,
                              float* agg, int E, int F) {
    size_t idx = (size_t)blockIdx.x * blockDim.x + threadIdx.x;
    size_t total = (size_t)E * F;
    if (idx >= total) return;
    int e = (int)(idx / F);
    int f = (int)(idx % F);
    int r = row[e], c = col[e];
    float w = dinv[r] * dinv[c];
    atomicAdd(&agg[(size_t)c * F + f], g[(size_t)r * F + f] * w);
}

// h = ReLU(agg + g*dinv^2 (self loop) + bias); write f32 and/or bf16
__global__ void finish_layer(const float* __restrict__ agg, const float* __restrict__ g,
                             const float* __restrict__ dinv, const float* __restrict__ bias,
                             float* outF, __bf16* outH, int Nn, int F) {
    size_t idx = (size_t)blockIdx.x * blockDim.x + threadIdx.x;
    size_t total = (size_t)Nn * F;
    if (idx >= total) return;
    int i = (int)(idx / F);
    int f = (int)(idx % F);
    float di = dinv[i];
    float v = fmaxf(agg[idx] + g[idx] * di * di + bias[f], 0.0f);
    if (outH) outH[idx] = f2bf(v);
    if (outF) outF[idx] = v;
}

// ---------------- small dense layers ----------------
__global__ void mlp_12_512(const float* __restrict__ h2, const float* __restrict__ L1,
                           const float* __restrict__ lb1, __bf16* __restrict__ h3, int Nn) {
    __shared__ float hs[12];
    int node = blockIdx.x;
    int j = threadIdx.x;                // blockDim = 512
    if (j < 12) hs[j] = h2[(size_t)node * 12 + j];
    __syncthreads();
    float acc = lb1[j];
    #pragma unroll
    for (int k = 0; k < 12; ++k) acc = fmaf(hs[k], L1[k * 512 + j], acc);
    h3[(size_t)node * 512 + j] = f2bf(fmaxf(acc, 0.0f));
}

// out[N x 2] = h4[N x 512] @ L3[512 x 2] + lb3. One wave32 per node.
__global__ void final_head(const __bf16* __restrict__ h4, const float* __restrict__ L3,
                           const float* __restrict__ lb3, float* __restrict__ out, int Nn) {
    int lane = threadIdx.x & 31;
    int node = blockIdx.x * (blockDim.x >> 5) + (threadIdx.x >> 5);
    if (node >= Nn) return;
    const __bf16* hp = h4 + (size_t)node * 512;
    float s0 = 0.f, s1 = 0.f;
    #pragma unroll
    for (int i = lane; i < 512; i += 32) {
        float h = (float)hp[i];
        s0 = fmaf(h, L3[i * 2 + 0], s0);
        s1 = fmaf(h, L3[i * 2 + 1], s1);
    }
    #pragma unroll
    for (int off = 16; off > 0; off >>= 1) {
        s0 += __shfl_down(s0, off, 32);
        s1 += __shfl_down(s1, off, 32);
    }
    if (lane == 0) {
        out[(size_t)node * 2 + 0] = s0 + lb3[0];
        out[(size_t)node * 2 + 1] = s1 + lb3[1];
    }
}

// ---------------- launcher ----------------
extern "C" void kernel_launch(void* const* d_in, const int* in_sizes, int n_in,
                              void* d_out, int out_size, void* d_ws, size_t ws_size,
                              hipStream_t stream) {
    const float* x   = (const float*)d_in[0];
    const int*  eidx = (const int*)  d_in[1];
    const float* W1  = (const float*)d_in[2];
    const float* b1  = (const float*)d_in[3];
    const float* W2  = (const float*)d_in[4];
    const float* b2  = (const float*)d_in[5];
    const float* L1  = (const float*)d_in[6];
    const float* lb1 = (const float*)d_in[7];
    const float* L2  = (const float*)d_in[8];
    const float* lb2 = (const float*)d_in[9];
    const float* L3  = (const float*)d_in[10];
    const float* lb3 = (const float*)d_in[11];
    float* out = (float*)d_out;
    (void)in_sizes; (void)n_in; (void)out_size; (void)ws_size;

    const int N = NNODES, E = NEDGES;
    const int* row = eidx;        // edge_index[0]
    const int* col = eidx + E;    // edge_index[1]

    char* ws = (char*)d_ws;
    size_t off = 0;
    auto alloc = [&](size_t bytes) -> void* {
        void* p = ws + off;
        off = (off + bytes + 255) & ~(size_t)255;
        return p;
    };
    float*  deg  = (float*) alloc((size_t)N * 4);
    float*  dinv = (float*) alloc((size_t)N * 4);
    __bf16* xbf  = (__bf16*)alloc((size_t)N * 256 * 2);
    __bf16* W1pk = (__bf16*)alloc((size_t)8 * 8 * 512 * 2);    // 128/16 x 256/32 tiles
    __bf16* W2pk = (__bf16*)alloc((size_t)1 * 4 * 512 * 2);    // 16/16  x 128/32
    __bf16* L2pk = (__bf16*)alloc((size_t)32 * 16 * 512 * 2);  // 512/16 x 512/32
    float*  g1   = (float*) alloc((size_t)N * 128 * 4);
    float*  agg1 = (float*) alloc((size_t)N * 128 * 4);
    __bf16* h1bf = (__bf16*)alloc((size_t)N * 128 * 2);
    float*  g2   = (float*) alloc((size_t)N * 12 * 4);
    float*  agg2 = (float*) alloc((size_t)N * 12 * 4);
    float*  h2   = (float*) alloc((size_t)N * 12 * 4);
    __bf16* h3bf = (__bf16*)alloc((size_t)N * 512 * 2);
    __bf16* h4bf = (__bf16*)alloc((size_t)N * 512 * 2);

    const int T = 256;
    // normalization
    deg_init <<<(N + T - 1) / T, T, 0, stream>>>(deg, N);
    deg_count<<<(E + T - 1) / T, T, 0, stream>>>(col, deg, E);
    deg_rsqrt<<<(N + T - 1) / T, T, 0, stream>>>(deg, dinv, N);
    // bf16 operands
    {
        size_t nx = (size_t)N * 256;
        f32_to_bf16<<<(unsigned)((nx + T - 1) / T), T, 0, stream>>>(x, xbf, nx);
    }
    { int tot = 8 * 8 * 512;   pack_B<<<(tot + T - 1) / T, T, 0, stream>>>(W1, W1pk, 256, 128, 128); }
    { int tot = 1 * 4 * 512;   pack_B<<<(tot + T - 1) / T, T, 0, stream>>>(W2, W2pk, 128, 12, 16); }
    { int tot = 32 * 16 * 512; pack_B<<<(tot + T - 1) / T, T, 0, stream>>>(L2, L2pk, 512, 512, 512); }

    const int mtiles  = N / 16;            // 3125, exact
    const int mgroups = (mtiles + 7) / 8;  // 391

    // GCN layer 1: g1 = x @ W1 (WMMA 16x64/wave, B via async LDS)  K=256 -> KT=8
    {
        const int ngroups = 128 / 64;      // NTB=4
        wmma_gemm_tiled<4, 0, 8><<<mgroups * ngroups, 256, 4 * 8 * 1024, stream>>>(
            xbf, W1pk, nullptr, g1, nullptr, N, 128, ngroups);
    }
    {
        size_t n1 = (size_t)N * 128;
        zero_f32<<<(unsigned)((n1 + T - 1) / T), T, 0, stream>>>(agg1, n1);
        size_t te = (size_t)E * 128;
        scatter_edges<<<(unsigned)((te + T - 1) / T), T, 0, stream>>>(row, col, dinv, g1, agg1, E, 128);
        finish_layer<<<(unsigned)((n1 + T - 1) / T), T, 0, stream>>>(agg1, g1, dinv, b1,
                                                                     nullptr, h1bf, N, 128);
    }
    // GCN layer 2: g2 = h1 @ W2 (Nout padded 12->16, NTB=1)  K=128 -> KT=4
    {
        wmma_gemm_tiled<1, 0, 4><<<mgroups * 1, 256, 1 * 4 * 1024, stream>>>(
            h1bf, W2pk, nullptr, g2, nullptr, N, 12, 1);
        size_t n2 = (size_t)N * 12;
        zero_f32<<<(unsigned)((n2 + T - 1) / T), T, 0, stream>>>(agg2, n2);
        size_t te = (size_t)E * 12;
        scatter_edges<<<(unsigned)((te + T - 1) / T), T, 0, stream>>>(row, col, dinv, g2, agg2, E, 12);
        finish_layer<<<(unsigned)((n2 + T - 1) / T), T, 0, stream>>>(agg2, g2, dinv, b2,
                                                                     h2, nullptr, N, 12);
    }
    // MLP head
    mlp_12_512<<<N, 512, 0, stream>>>(h2, L1, lb1, h3bf, N);
    {
        const int ngroups = 512 / 64;      // dominant GEMM: 50000x512x512, K=512 -> KT=16
        wmma_gemm_tiled<4, 1, 16><<<mgroups * ngroups, 256, 4 * 16 * 1024, stream>>>(
            h3bf, L2pk, lb2, nullptr, h4bf, N, 512, ngroups);
    }
    final_head<<<(N + 7) / 8, 256, 0, stream>>>(h4bf, L3, lb3, out, N);
}
